// MHAtt2D_12781822673061
// MI455X (gfx1250) — compile-verified, matrix operand order checked by reading, not run on hardware
//
#include <hip/hip_runtime.h>

// ---------------------------------------------------------------------------
// MHAtt2D for MI455X (gfx1250): all GEMM work on v_wmma_f32_16x16x32_bf16.
// B=8, L=HW=1024, C=512, nh=8, dh=64.
//
// bf16 WMMA fragment layout (wave32, 16x16x32):
//   A (16x32): lane l holds row m=l%16, K chunks [h8..h8+7] and [16+h8..16+h8+7]
//              where h8 = (l/16)*8  -> two contiguous 16B chunks of a row.
//   B (32x16): same pattern applied to rows of B-transposed.
// So both operands load as 2x b128 per lane from row-major [*,K] storage.
//
// v2 changes: weights pre-converted to bf16 once (kills redundant cvt+bytes),
// 16x64 strips per wave in the GEMM kernels (4x A-frag reuse), 32-query tiles
// in attention (2x B-frag reuse), global_prefetch hints on streaming reads.
// ---------------------------------------------------------------------------

typedef __attribute__((ext_vector_type(16))) __bf16 v16bf;
typedef __attribute__((ext_vector_type(8)))  __bf16 bf16x8;
typedef __attribute__((ext_vector_type(4)))  __bf16 bf16x4;
typedef __attribute__((ext_vector_type(8)))  float  f32x8;

#define NEG_INF_F (-1.0e9f)

__device__ __forceinline__ f32x8 wmma_bf16(v16bf a, v16bf b, f32x8 c) {
  return __builtin_amdgcn_wmma_f32_16x16x32_bf16(
      /*neg_a=*/false, a, /*neg_b=*/false, b,
      /*c_mod=*/(short)0, c, /*reuse_a=*/false, /*reuse_b=*/false);
}

// fragment from a row-major f32 matrix row (convert f32 -> bf16 in flight)
__device__ __forceinline__ v16bf frag_f32(const float* __restrict__ row, int koff) {
  const float4* p0 = reinterpret_cast<const float4*>(row + koff);
  const float4* p1 = reinterpret_cast<const float4*>(row + koff + 16);
  float4 a = p0[0], b = p0[1], c = p1[0], d = p1[1];
  v16bf f;
  f[0]  = (__bf16)a.x; f[1]  = (__bf16)a.y; f[2]  = (__bf16)a.z; f[3]  = (__bf16)a.w;
  f[4]  = (__bf16)b.x; f[5]  = (__bf16)b.y; f[6]  = (__bf16)b.z; f[7]  = (__bf16)b.w;
  f[8]  = (__bf16)c.x; f[9]  = (__bf16)c.y; f[10] = (__bf16)c.z; f[11] = (__bf16)c.w;
  f[12] = (__bf16)d.x; f[13] = (__bf16)d.y; f[14] = (__bf16)d.z; f[15] = (__bf16)d.w;
  return f;
}

// fragment from a row-major bf16 matrix row (two 16B b128 loads)
__device__ __forceinline__ v16bf frag_bf16(const __bf16* __restrict__ row, int koff) {
  bf16x8 c0 = *reinterpret_cast<const bf16x8*>(row + koff);
  bf16x8 c1 = *reinterpret_cast<const bf16x8*>(row + koff + 16);
  v16bf f;
#pragma unroll
  for (int j = 0; j < 8; ++j) { f[j] = c0[j]; f[j + 8] = c1[j]; }
  return f;
}

// ---------------------------------------------------------------------------
// Kernel 0: one-shot f32 -> bf16 weight conversion (512x512 = 65536 float4).
// ---------------------------------------------------------------------------
__global__ void __launch_bounds__(256)
cvt_w_kernel(const float* __restrict__ src, __bf16* __restrict__ dst) {
  const int i = blockIdx.x * 256 + threadIdx.x;   // exactly 65536 threads
  float4 v = reinterpret_cast<const float4*>(src)[i];
  bf16x4 p;
  p[0] = (__bf16)v.x; p[1] = (__bf16)v.y; p[2] = (__bf16)v.z; p[3] = (__bf16)v.w;
  reinterpret_cast<bf16x4*>(dst)[i] = p;
}

// ---------------------------------------------------------------------------
// Kernel 1: K and V linear projections.  X[8192,512] @ W^T + b.
// One wave computes a 16(M) x 64(N) strip (one head) -> A-frag reused 4x.
//   z==0 -> K:  write kh[b][h][l][dh]   (row-major over dh)
//   z==1 -> V:  write vhT[b][h][dh][l]  (transposed: row-major over l)
// ---------------------------------------------------------------------------
__global__ void __launch_bounds__(128)
proj_kv_kernel(const float* __restrict__ vin, const float* __restrict__ kin,
               const __bf16* __restrict__ Wvb, const float* __restrict__ bv,
               const __bf16* __restrict__ Wkb, const float* __restrict__ bk,
               __bf16* __restrict__ kh, __bf16* __restrict__ vhT) {
  const int lane  = threadIdx.x;
  const int row16 = lane & 15;
  const int hlf   = lane >> 4;
  const int rt = blockIdx.x;                                // 0..511
  const int st = blockIdx.y * blockDim.y + threadIdx.y;     // head strip 0..7
  const bool isV = (blockIdx.z != 0);
  const float*  X    = isV ? vin : kin;
  const __bf16* Wb   = isV ? Wvb : Wkb;
  const float*  bias = isV ? bv  : bk;

  const int r0 = rt * 16, o0 = st * 64;
  const int arow = r0 + row16;

  f32x8 acc[4] = {};
#pragma unroll 2
  for (int ks = 0; ks < 16; ++ks) {
    const int koff = ks * 32 + hlf * 8;
    v16bf a = frag_f32(X + (size_t)arow * 512, koff);
#pragma unroll
    for (int nt = 0; nt < 4; ++nt) {
      const __bf16* wrow = Wb + (size_t)(o0 + nt * 16 + row16) * 512;
      acc[nt] = wmma_bf16(a, frag_bf16(wrow, koff), acc[nt]);
    }
  }

  const int bb = r0 >> 10, l0 = r0 & 1023;
  const int h = st;                                         // strip == head
  if (!isV) {
    __bf16* dst = kh + ((size_t)bb * 8 + h) * 1024 * 64;
#pragma unroll
    for (int nt = 0; nt < 4; ++nt) {
      const int dh = nt * 16 + row16;
      const float bia = bias[o0 + nt * 16 + row16];
#pragma unroll
      for (int r = 0; r < 8; ++r) {
        const int l = l0 + r + 8 * hlf;
        dst[(size_t)l * 64 + dh] = (__bf16)(acc[nt][r] + bia);
      }
    }
  } else {
    __bf16* dst = vhT + ((size_t)bb * 8 + h) * 64 * 1024;
#pragma unroll
    for (int nt = 0; nt < 4; ++nt) {
      const int dh = nt * 16 + row16;
      const float bia = bias[o0 + nt * 16 + row16];
      bf16x8 pk;
#pragma unroll
      for (int r = 0; r < 8; ++r) pk[r] = (__bf16)(acc[nt][r] + bia);
      *reinterpret_cast<bf16x8*>(dst + (size_t)dh * 1024 + l0 + 8 * hlf) = pk;
    }
  }
}

// ---------------------------------------------------------------------------
// Kernel 2: Q 1x1-conv projection. X[i,c] = q[b, c, hw] (channel-strided
// gather, converted to bf16 in flight); 16x64 strip per wave.
// ---------------------------------------------------------------------------
__global__ void __launch_bounds__(128)
proj_q_kernel(const float* __restrict__ qin, const __bf16* __restrict__ Wqb,
              const float* __restrict__ bq, __bf16* __restrict__ qh) {
  const int lane  = threadIdx.x;
  const int row16 = lane & 15;
  const int hlf   = lane >> 4;
  const int rt = blockIdx.x;                                // 0..511
  const int st = blockIdx.y * blockDim.y + threadIdx.y;     // 0..7

  const int r0 = rt * 16, o0 = st * 64;
  const int bb = r0 >> 10, hw0 = r0 & 1023;
  const float* qbase = qin + (size_t)bb * 512 * 1024 + (hw0 + row16);

  f32x8 acc[4] = {};
#pragma unroll 2
  for (int ks = 0; ks < 16; ++ks) {
    const int koff = ks * 32 + hlf * 8;
    v16bf a;
#pragma unroll
    for (int j = 0; j < 8; ++j) {
      a[j]     = (__bf16)qbase[(size_t)(koff + j) * 1024];
      a[j + 8] = (__bf16)qbase[(size_t)(koff + 16 + j) * 1024];
    }
#pragma unroll
    for (int nt = 0; nt < 4; ++nt) {
      const __bf16* wrow = Wqb + (size_t)(o0 + nt * 16 + row16) * 512;
      acc[nt] = wmma_bf16(a, frag_bf16(wrow, koff), acc[nt]);
    }
  }

  const int h = st;
  __bf16* dst = qh + ((size_t)bb * 8 + h) * 1024 * 64;
#pragma unroll
  for (int nt = 0; nt < 4; ++nt) {
    const int dh = nt * 16 + row16;
    const float bia = bq[o0 + nt * 16 + row16];
#pragma unroll
    for (int r = 0; r < 8; ++r) {
      const int hw = hw0 + r + 8 * hlf;
      dst[(size_t)hw * 64 + dh] = (__bf16)(acc[nt][r] + bia);
    }
  }
}

// ---------------------------------------------------------------------------
// Kernel 3: attention for one (b, head, 32-query tile); one wave per block.
//   pass 1: S = (Q K^T)/8 + mask -> bf16 scores in 64KB LDS (K/V frags
//           shared by the two 16-row A tiles -> 4 wmma per 2 key loads)
//   row-max scan (one lane per row), pass 2: P=exp(S-max), O += P*vhT,
//   normalize by rowsum, store attout[b][q][h*64+dh].
// ---------------------------------------------------------------------------
__global__ void __launch_bounds__(32)
attn_kernel(const __bf16* __restrict__ qh, const __bf16* __restrict__ kh,
            const __bf16* __restrict__ vhT, const unsigned char* __restrict__ mask,
            __bf16* __restrict__ attout) {
  __shared__ __align__(16) __bf16 sS[32 * 1024];            // 64 KB scores
  __shared__ float sMax[32];
  __shared__ float sSum[32];

  const int lane  = threadIdx.x;
  const int row16 = lane & 15;
  const int hlf   = lane >> 4;
  const int qt = blockIdx.x;                                // 0..31
  const int h  = blockIdx.y;                                // 0..7
  const int b  = blockIdx.z;                                // 0..7
  const size_t bh = (size_t)b * 8 + h;
  const int q0 = qt * 32;

  // Q fragments (two 16-row tiles), loop-invariant
  const __bf16* qrow0 = qh + (bh * 1024 + q0 + row16) * 64;
  const __bf16* qrow1 = qrow0 + 16 * 64;
  const v16bf a00 = frag_bf16(qrow0, hlf * 8);
  const v16bf a01 = frag_bf16(qrow0, 32 + hlf * 8);
  const v16bf a10 = frag_bf16(qrow1, hlf * 8);
  const v16bf a11 = frag_bf16(qrow1, 32 + hlf * 8);

  const __bf16* kbase = kh + bh * 1024 * 64;
  const unsigned char* mrow = mask + (size_t)b * 1024;

  // ---- pass 1: score tiles -> LDS (masked, scaled 1/sqrt(64)) ----
  for (int kt = 0; kt < 64; ++kt) {
    const int key = kt * 16 + row16;                        // lane's N column
    const __bf16* krow = kbase + (size_t)key * 64;
    if (kt < 63) __builtin_prefetch(krow + 16 * 64, 0, 1);  // next key tile
    v16bf b0 = frag_bf16(krow, hlf * 8);
    v16bf b1 = frag_bf16(krow, 32 + hlf * 8);
    f32x8 s0 = {}, s1 = {};
    s0 = wmma_bf16(a00, b0, s0);
    s0 = wmma_bf16(a01, b1, s0);
    s1 = wmma_bf16(a10, b0, s1);
    s1 = wmma_bf16(a11, b1, s1);
    const bool m = (mrow[key] != 0);
#pragma unroll
    for (int r = 0; r < 8; ++r) {
      sS[(size_t)(r + 8 * hlf) * 1024 + key]      = (__bf16)(m ? NEG_INF_F : s0[r] * 0.125f);
      sS[(size_t)(16 + r + 8 * hlf) * 1024 + key] = (__bf16)(m ? NEG_INF_F : s1[r] * 0.125f);
    }
  }
  __syncthreads();

  // ---- row max: lane l scans row l ----
  {
    float mx = -3.0e38f;
    const __bf16* rs = &sS[(size_t)lane * 1024];
#pragma unroll 4
    for (int i = 0; i < 1024; i += 8) {
      bf16x8 v = *reinterpret_cast<const bf16x8*>(rs + i);
#pragma unroll
      for (int j = 0; j < 8; ++j) mx = fmaxf(mx, (float)v[j]);
    }
    sMax[lane] = mx;
  }
  __syncthreads();

  // ---- pass 2: O = exp(S-max) * V  (32 keys per iteration, 8 wmma) ----
  f32x8 o[2][4] = {};
  const float rmax0 = sMax[row16];
  const float rmax1 = sMax[16 + row16];
  float rsum0 = 0.0f, rsum1 = 0.0f;
  const __bf16* vbase = vhT + bh * 64 * 1024;
  const __bf16* pr0 = &sS[(size_t)row16 * 1024];
  const __bf16* pr1 = &sS[(size_t)(16 + row16) * 1024];
  for (int kt = 0; kt < 32; ++kt) {
    const int kb = kt * 32 + hlf * 8;
    if (kt < 31) __builtin_prefetch(vbase + (size_t)row16 * 1024 + kb + 32, 0, 1);
    bf16x8 x0 = *reinterpret_cast<const bf16x8*>(pr0 + kb);
    bf16x8 x1 = *reinterpret_cast<const bf16x8*>(pr0 + kb + 16);
    bf16x8 y0 = *reinterpret_cast<const bf16x8*>(pr1 + kb);
    bf16x8 y1 = *reinterpret_cast<const bf16x8*>(pr1 + kb + 16);
    v16bf aP0, aP1;
#pragma unroll
    for (int j = 0; j < 8; ++j) {
      float e0 = __expf((float)x0[j] - rmax0);
      float e1 = __expf((float)x1[j] - rmax0);
      float f0 = __expf((float)y0[j] - rmax1);
      float f1 = __expf((float)y1[j] - rmax1);
      rsum0 += e0 + e1;
      rsum1 += f0 + f1;
      aP0[j] = (__bf16)e0; aP0[j + 8] = (__bf16)e1;
      aP1[j] = (__bf16)f0; aP1[j + 8] = (__bf16)f1;
    }
#pragma unroll
    for (int nt = 0; nt < 4; ++nt) {
      const __bf16* vrow = vbase + (size_t)(nt * 16 + row16) * 1024;
      v16bf bf = frag_bf16(vrow, kb);
      o[0][nt] = wmma_bf16(aP0, bf, o[0][nt]);
      o[1][nt] = wmma_bf16(aP1, bf, o[1][nt]);
    }
  }
  rsum0 += __shfl_xor(rsum0, 16, 32);
  rsum1 += __shfl_xor(rsum1, 16, 32);
  if (hlf == 0) { sSum[row16] = rsum0; sSum[16 + row16] = rsum1; }
  __syncthreads();

  // ---- normalize + store attout[b][q][h*64+dh] (bf16 for the out-proj) ----
#pragma unroll
  for (int t = 0; t < 2; ++t) {
    __bf16* orow = attout + ((size_t)b * 1024 + q0 + t * 16) * 512 + h * 64;
#pragma unroll
    for (int nt = 0; nt < 4; ++nt) {
      const int dh = nt * 16 + row16;
#pragma unroll
      for (int r = 0; r < 8; ++r) {
        const int M = r + 8 * hlf;
        orow[(size_t)M * 512 + dh] = (__bf16)(o[t][nt][r] / sSum[t * 16 + M]);
      }
    }
  }
}

// ---------------------------------------------------------------------------
// Kernel 4: output projection + bias + ReLU; 16x64 strip per wave.
// Y[i,o] = relu(X @ Wm^T + bm), written as out[b][o][hw] f32 (b128 stores).
// ---------------------------------------------------------------------------
__global__ void __launch_bounds__(128)
proj_out_kernel(const __bf16* __restrict__ attout, const __bf16* __restrict__ Wmb,
                const float* __restrict__ bm, float* __restrict__ out) {
  const int lane  = threadIdx.x;
  const int row16 = lane & 15;
  const int hlf   = lane >> 4;
  const int rt = blockIdx.x;                                // 0..511
  const int st = blockIdx.y * blockDim.y + threadIdx.y;     // 0..7

  const int r0 = rt * 16, o0 = st * 64;
  const int arow = r0 + row16;

  f32x8 acc[4] = {};
#pragma unroll 2
  for (int ks = 0; ks < 16; ++ks) {
    const int koff = ks * 32 + hlf * 8;
    v16bf a = frag_bf16(attout + (size_t)arow * 512, koff);
#pragma unroll
    for (int nt = 0; nt < 4; ++nt) {
      const __bf16* wrow = Wmb + (size_t)(o0 + nt * 16 + row16) * 512;
      acc[nt] = wmma_bf16(a, frag_bf16(wrow, koff), acc[nt]);
    }
  }

  const int bb = r0 >> 10, hw0 = r0 & 1023;
#pragma unroll
  for (int nt = 0; nt < 4; ++nt) {
    const int ocol = o0 + nt * 16 + row16;
    const float bia = bm[ocol];
    float* dst = out + ((size_t)bb * 512 + ocol) * 1024 + hw0 + 8 * hlf;
    float4 s0 = make_float4(fmaxf(acc[nt][0] + bia, 0.0f), fmaxf(acc[nt][1] + bia, 0.0f),
                            fmaxf(acc[nt][2] + bia, 0.0f), fmaxf(acc[nt][3] + bia, 0.0f));
    float4 s1 = make_float4(fmaxf(acc[nt][4] + bia, 0.0f), fmaxf(acc[nt][5] + bia, 0.0f),
                            fmaxf(acc[nt][6] + bia, 0.0f), fmaxf(acc[nt][7] + bia, 0.0f));
    reinterpret_cast<float4*>(dst)[0] = s0;
    reinterpret_cast<float4*>(dst)[1] = s1;
  }
}

// ---------------------------------------------------------------------------
// Launch
// ---------------------------------------------------------------------------
extern "C" void kernel_launch(void* const* d_in, const int* in_sizes, int n_in,
                              void* d_out, int out_size, void* d_ws, size_t ws_size,
                              hipStream_t stream) {
  const float* vin = (const float*)d_in[0];
  const float* kin = (const float*)d_in[1];
  const float* qin = (const float*)d_in[2];
  const unsigned char* mask = (const unsigned char*)d_in[3];
  const float* Wv = (const float*)d_in[4];
  const float* bv = (const float*)d_in[5];
  const float* Wk = (const float*)d_in[6];
  const float* bk = (const float*)d_in[7];
  const float* Wq = (const float*)d_in[8];
  const float* bq = (const float*)d_in[9];
  const float* Wm = (const float*)d_in[10];
  const float* bm = (const float*)d_in[11];
  float* out = (float*)d_out;

  // workspace: qh/kh/vhT/attout (4 x 8MB bf16) + 4 bf16 weight copies (2MB)
  char* ws = (char*)d_ws;
  __bf16* qh     = (__bf16*)(ws);
  __bf16* kh     = (__bf16*)(ws + (size_t)(8u << 20));
  __bf16* vhT    = (__bf16*)(ws + (size_t)(16u << 20));
  __bf16* attout = (__bf16*)(ws + (size_t)(24u << 20));
  __bf16* Wkb    = (__bf16*)(ws + (size_t)(32u << 20));
  __bf16* Wvb    = Wkb + 512 * 512;
  __bf16* Wqb    = Wvb + 512 * 512;
  __bf16* Wmb    = Wqb + 512 * 512;

  cvt_w_kernel<<<256, 256, 0, stream>>>(Wk, Wkb);
  cvt_w_kernel<<<256, 256, 0, stream>>>(Wv, Wvb);
  cvt_w_kernel<<<256, 256, 0, stream>>>(Wq, Wqb);
  cvt_w_kernel<<<256, 256, 0, stream>>>(Wm, Wmb);

  const dim3 blk(32, 4);
  proj_kv_kernel<<<dim3(512, 2, 2), blk, 0, stream>>>(vin, kin, Wvb, bv, Wkb, bk, kh, vhT);
  proj_q_kernel<<<dim3(512, 2, 1), blk, 0, stream>>>(qin, Wqb, bq, qh);
  attn_kernel<<<dim3(32, 8, 8), dim3(32, 1, 1), 0, stream>>>(qh, kh, vhT, mask, attout);
  proj_out_kernel<<<dim3(512, 2, 1), blk, 0, stream>>>(attout, Wmb, bm, out);
}
